// MoeLayer_41583873360109
// MI455X (gfx1250) — matrix operand order repaired and checked by reading, not compiled, not run on hardware
//
#include <hip/hip_runtime.h>
#include <hip/hip_bf16.h>
#include <stdint.h>

// ---------------------------------------------------------------------------
// MoE SwiGLU for MI455X (gfx1250, wave32, WMMA).
//   B=4 S=1024 D=1024 E=8 K=2 H=2048, T = B*S = 4096 tokens.
// Routed (top-2) grouped GEMMs, bf16 WMMA with f32 accumulate:
//   103 GFLOP routed vs 412 GFLOP dense (4x algorithmic win).
// Weights & activations pre-converted f32->bf16 once per launch; GEMM inner
// loops do zero conversion work. BM=32: 2 A fragments/wave -> 4 WMMAs/k-step.
// A-tile staging is register-pipelined one iteration ahead (issue after the
// barrier, consume next iteration) -> no synchronous global->LDS bubble, one
// barrier per k-step. k-loops at unroll 1 keep accumulators in place.
// Workspace layout (requires ws_size >= ~177 MB):
//   [cnt][rec][wslot][x_bf16][W1_bf16][W2_bf16][W3_bf16][g bf16][y f32]
// ---------------------------------------------------------------------------

typedef __attribute__((ext_vector_type(16))) __bf16 v16bf;
typedef __attribute__((ext_vector_type(8)))  __bf16 v8bf;
typedef __attribute__((ext_vector_type(8)))  float  v8f;

#define D_    1024
#define H_    2048
#define E_    8
#define T_    4096
#define TCAP  4096            // worst-case tokens per expert
#define BM    32              // token rows per block tile (2 WMMA M-tiles)
#define KSTEP 32              // WMMA K per step (bf16 16x16x32)
#define BLK   256             // 8 waves (wave32)
#define WAVES 8

// ---------------------------------------------------------------- init ------
__global__ void moe_init(int* __restrict__ cnt, int* __restrict__ rec,
                         float* __restrict__ wslot) {
    int idx = blockIdx.x * blockDim.x + threadIdx.x;
    if (idx < E_) cnt[idx] = 0;
    for (int i = idx; i < E_ * TCAP; i += gridDim.x * blockDim.x) {
        rec[i]   = 2 * T_;   // pad slot -> dummy g/y row
        wslot[i] = 0.0f;
    }
}

// ------------------------------------------------------- f32 -> bf16 --------
__global__ __launch_bounds__(256) void cvt_f32_bf16(
    const float* __restrict__ src, __bf16* __restrict__ dst) {
    size_t i = (size_t)blockIdx.x * blockDim.x + threadIdx.x;  // 8 elem/thread
    const float4* s4 = (const float4*)src;
    float4 a = s4[i * 2], b = s4[i * 2 + 1];
    v8bf o;
    o[0] = (__bf16)a.x; o[1] = (__bf16)a.y; o[2] = (__bf16)a.z; o[3] = (__bf16)a.w;
    o[4] = (__bf16)b.x; o[5] = (__bf16)b.y; o[6] = (__bf16)b.z; o[7] = (__bf16)b.w;
    *(v8bf*)(dst + i * 8) = o;
}

// ---------------------------------------------------------------- gating ----
__global__ __launch_bounds__(128) void moe_gate(
    const float* __restrict__ x, const float* __restrict__ Wg,
    int* __restrict__ cnt, int* __restrict__ rec, float* __restrict__ wslot) {
    __shared__ __align__(16) float wg[E_ * D_];     // 32 KB
    const int tid = threadIdx.x;

    const float4* wg4g = (const float4*)Wg;
    float4* wg4s = (float4*)wg;
    for (int i = tid; i < E_ * D_ / 4; i += 128) wg4s[i] = wg4g[i];
    __syncthreads();

    const int t = blockIdx.x * 128 + tid;
    const float4* x4 = (const float4*)(x + (size_t)t * D_);
    float acc[E_];
#pragma unroll
    for (int e = 0; e < E_; e++) acc[e] = 0.0f;
    for (int i = 0; i < D_ / 4; i++) {
        float4 xv = x4[i];
#pragma unroll
        for (int e = 0; e < E_; e++) {
            float4 wv = ((const float4*)(wg + e * D_))[i];
            acc[e] += xv.x * wv.x + xv.y * wv.y + xv.z * wv.z + xv.w * wv.w;
        }
    }
    int i0 = 0; float v0 = acc[0];
#pragma unroll
    for (int e = 1; e < E_; e++) if (acc[e] > v0) { v0 = acc[e]; i0 = e; }
    int i1 = (i0 == 0) ? 1 : 0; float v1 = acc[i1];
#pragma unroll
    for (int e = 0; e < E_; e++)
        if (e != i0 && acc[e] > v1) { v1 = acc[e]; i1 = e; }

    float e1 = __expf(v1 - v0);
    float r  = __builtin_amdgcn_rcpf(1.0f + e1);
    float p0 = r;
    float p1 = e1 * r;

    int pos0 = atomicAdd(&cnt[i0], 1);
    rec[i0 * TCAP + pos0] = t * 2;     wslot[i0 * TCAP + pos0] = p0;
    int pos1 = atomicAdd(&cnt[i1], 1);
    rec[i1 * TCAP + pos1] = t * 2 + 1; wslot[i1 * TCAP + pos1] = p1;
}

// ------------------------------------------------- FFN stage 1 (x->g) -------
// h1 = x@W1[e]^T, h2 = x@W2[e]^T, g = h1*sigmoid(h1)*h2  (fused epilogue)
// BM=32 rows, 8 waves x 16 cols: 4 accumulators/wave (h1,h2 x 2 row tiles).
__global__ __launch_bounds__(BLK) void moe_ffn1(
    const __bf16* __restrict__ xb, const __bf16* __restrict__ W1b,
    const __bf16* __restrict__ W2b, const int* __restrict__ cnt,
    const int* __restrict__ rec, __bf16* __restrict__ g) {
    const int mtiles = TCAP / BM;                 // 128
    const int e  = blockIdx.x / mtiles;
    const int mt = blockIdx.x % mtiles;
    const int c  = cnt[e];
    if (mt * BM >= c) return;                     // uniform early exit
    const int n0   = blockIdx.y * (WAVES * 16);   // 128
    const int wave = threadIdx.x >> 5;
    const int lane = threadIdx.x & 31;
    const int nw   = n0 + wave * 16;

    __shared__ __align__(16) __bf16 As[2][BM][KSTEP];   // 4 KB ping-pong
    __shared__ int rowS[BM];
    __shared__ int tokS[BM];

    if (threadIdx.x < BM) {
        int r = rec[e * TCAP + mt * BM + threadIdx.x];
        rowS[threadIdx.x] = r;
        int t = r >> 1;
        tokS[threadIdx.x] = (t < T_) ? t : (T_ - 1);   // pad-safe gather
    }
    __syncthreads();

    const int am = lane & 15;          // A row within tile (M)
    const int kh = lane >> 4;          // K half selector
    const __bf16* b1p = W1b + ((size_t)e * H_ + nw + (lane & 15)) * (size_t)D_ + kh * 16;
    const __bf16* b2p = W2b + ((size_t)e * H_ + nw + (lane & 15)) * (size_t)D_ + kh * 16;

    const int ldrow = (threadIdx.x * 4) >> 5;    // 4 bf16 (8B) per thread
    const int ldcol = (threadIdx.x * 4) & 31;
    const __bf16* xrow = xb + (size_t)tokS[ldrow] * D_ + ldcol;

    v8f c1a = {}, c1b = {}, c2a = {}, c2b = {};

    // register-pipelined staging: holds tile ks at top of iteration ks
    uint2 stage = *(const uint2*)(xrow);

    const int NK = D_ / KSTEP;                    // 32
#pragma unroll 1
    for (int ks = 0; ks < NK; ks++) {
        const int buf = ks & 1;
        *(uint2*)&As[buf][ldrow][ldcol] = stage;
        __syncthreads();
        // issue next tile's load now; consumed next iteration (ks=NK-1 reads
        // <=64B past the row into mapped workspace and is never consumed).
        stage = *(const uint2*)(xrow + (ks + 1) * KSTEP);

        // A fragments: rows [0,16) and [16,32)
        v16bf a0, a1;
        {
            v8bf lo = *(const v8bf*)&As[buf][am][kh * 16];
            v8bf hi = *(const v8bf*)&As[buf][am][kh * 16 + 8];
#pragma unroll
            for (int i = 0; i < 8; i++) { a0[i] = lo[i]; a0[i + 8] = hi[i]; }
        }
        {
            v8bf lo = *(const v8bf*)&As[buf][16 + am][kh * 16];
            v8bf hi = *(const v8bf*)&As[buf][16 + am][kh * 16 + 8];
#pragma unroll
            for (int i = 0; i < 8; i++) { a1[i] = lo[i]; a1[i + 8] = hi[i]; }
        }

        // B fragments: 16 K-contiguous bf16 per lane (32B, 2x b128 loads)
        v16bf b1 = *(const v16bf*)(b1p + ks * KSTEP);
        v16bf b2 = *(const v16bf*)(b2p + ks * KSTEP);

        // grouped by A fragment to keep live ranges short
        c1a = __builtin_amdgcn_wmma_f32_16x16x32_bf16(false, a0, false, b1,
                                                      (short)0, c1a, false, false);
        c2a = __builtin_amdgcn_wmma_f32_16x16x32_bf16(false, a0, false, b2,
                                                      (short)0, c2a, false, false);
        c1b = __builtin_amdgcn_wmma_f32_16x16x32_bf16(false, a1, false, b1,
                                                      (short)0, c1b, false, false);
        c2b = __builtin_amdgcn_wmma_f32_16x16x32_bf16(false, a1, false, b2,
                                                      (short)0, c2b, false, false);
    }

    // SwiGLU epilogue: g = h1 * sigmoid(h1) * h2  -> bf16 (fast rcp)
    const int ncol  = nw + (lane & 15);
    const int mbase = (lane >> 4) * 8;
#pragma unroll
    for (int i = 0; i < 8; i++) {
        float h1 = c1a[i], h2 = c2a[i];
        float gv = h1 * h2 * __builtin_amdgcn_rcpf(1.0f + __expf(-h1));
        g[(size_t)rowS[mbase + i] * H_ + ncol] = (__bf16)gv;

        float h1x = c1b[i], h2x = c2b[i];
        float gvx = h1x * h2x * __builtin_amdgcn_rcpf(1.0f + __expf(-h1x));
        g[(size_t)rowS[16 + mbase + i] * H_ + ncol] = (__bf16)gvx;
    }
}

// ------------------------------------------------- FFN stage 2 (g->y) -------
// y[slot] = (g[slot] @ W3[e]^T) * prob[slot]; BM=32 x BN=256 per block.
__global__ __launch_bounds__(BLK) void moe_ffn2(
    const __bf16* __restrict__ g, const __bf16* __restrict__ W3b,
    const int* __restrict__ cnt, const int* __restrict__ rec,
    const float* __restrict__ wslot, float* __restrict__ y) {
    const int mtiles = TCAP / BM;                 // 128
    const int e  = blockIdx.x / mtiles;
    const int mt = blockIdx.x % mtiles;
    const int c  = cnt[e];
    if (mt * BM >= c) return;
    const int n0   = blockIdx.y * (WAVES * 16 * 2);   // 256
    const int wave = threadIdx.x >> 5;
    const int lane = threadIdx.x & 31;
    const int nwa  = n0 + wave * 16;
    const int nwb  = nwa + WAVES * 16;                // +128

    __shared__ __align__(16) __bf16 As[2][BM][KSTEP];
    __shared__ int   rowS[BM];
    __shared__ float wS[BM];

    if (threadIdx.x < BM) {
        int idx = e * TCAP + mt * BM + threadIdx.x;
        rowS[threadIdx.x] = rec[idx];
        wS[threadIdx.x]   = wslot[idx];
    }
    __syncthreads();

    const int am = lane & 15;
    const int kh = lane >> 4;
    const __bf16* b3pa = W3b + ((size_t)e * D_ + nwa + (lane & 15)) * (size_t)H_ + kh * 16;
    const __bf16* b3pb = W3b + ((size_t)e * D_ + nwb + (lane & 15)) * (size_t)H_ + kh * 16;

    const int ldrow = (threadIdx.x * 4) >> 5;
    const int ldcol = (threadIdx.x * 4) & 31;
    const __bf16* grow_p = g + (size_t)rowS[ldrow] * H_ + ldcol;

    v8f caa = {}, cab = {}, cba = {}, cbb = {};   // c[row tile][col tile]

    uint2 stage = *(const uint2*)(grow_p);

    const int NK = H_ / KSTEP;                    // 64
#pragma unroll 1
    for (int ks = 0; ks < NK; ks++) {
        const int buf = ks & 1;
        *(uint2*)&As[buf][ldrow][ldcol] = stage;
        __syncthreads();
        stage = *(const uint2*)(grow_p + (ks + 1) * KSTEP);  // g has spare rows

        v16bf a0, a1;
        {
            v8bf lo = *(const v8bf*)&As[buf][am][kh * 16];
            v8bf hi = *(const v8bf*)&As[buf][am][kh * 16 + 8];
#pragma unroll
            for (int i = 0; i < 8; i++) { a0[i] = lo[i]; a0[i + 8] = hi[i]; }
        }
        {
            v8bf lo = *(const v8bf*)&As[buf][16 + am][kh * 16];
            v8bf hi = *(const v8bf*)&As[buf][16 + am][kh * 16 + 8];
#pragma unroll
            for (int i = 0; i < 8; i++) { a1[i] = lo[i]; a1[i + 8] = hi[i]; }
        }

        v16bf b3a = *(const v16bf*)(b3pa + ks * KSTEP);
        v16bf b3b = *(const v16bf*)(b3pb + ks * KSTEP);

        caa = __builtin_amdgcn_wmma_f32_16x16x32_bf16(false, a0, false, b3a,
                                                      (short)0, caa, false, false);
        cab = __builtin_amdgcn_wmma_f32_16x16x32_bf16(false, a0, false, b3b,
                                                      (short)0, cab, false, false);
        cba = __builtin_amdgcn_wmma_f32_16x16x32_bf16(false, a1, false, b3a,
                                                      (short)0, cba, false, false);
        cbb = __builtin_amdgcn_wmma_f32_16x16x32_bf16(false, a1, false, b3b,
                                                      (short)0, cbb, false, false);
    }

    const int ncola = nwa + (lane & 15);
    const int ncolb = nwb + (lane & 15);
    const int mbase = (lane >> 4) * 8;
#pragma unroll
    for (int i = 0; i < 8; i++) {
        int   r0 = rowS[mbase + i];
        float w0 = wS[mbase + i];
        y[(size_t)r0 * D_ + ncola] = caa[i] * w0;
        y[(size_t)r0 * D_ + ncolb] = cab[i] * w0;

        int   r1 = rowS[16 + mbase + i];
        float w1 = wS[16 + mbase + i];
        y[(size_t)r1 * D_ + ncola] = cba[i] * w1;
        y[(size_t)r1 * D_ + ncolb] = cbb[i] * w1;
    }
}

// ---------------------------------------------------------------- combine ---
// out[t] = y[2t] + y[2t+1] ; fully overwrites d_out, no atomics.
__global__ void moe_combine(const float* __restrict__ y, float* __restrict__ out) {
    int idx = blockIdx.x * blockDim.x + threadIdx.x;      // over T_*D_/4
    int t  = idx / (D_ / 4);
    int d4 = idx % (D_ / 4);
    const float4* y4 = (const float4*)y;
    float4 a = y4[(size_t)(2 * t) * (D_ / 4) + d4];
    float4 b = y4[(size_t)(2 * t + 1) * (D_ / 4) + d4];
    ((float4*)out)[idx] = make_float4(a.x + b.x, a.y + b.y, a.z + b.z, a.w + b.w);
}

// ---------------------------------------------------------------- launch ----
extern "C" void kernel_launch(void* const* d_in, const int* in_sizes, int n_in,
                              void* d_out, int out_size, void* d_ws, size_t ws_size,
                              hipStream_t stream) {
    const float* x  = (const float*)d_in[0];
    const float* Wg = (const float*)d_in[1];
    const float* W1 = (const float*)d_in[2];
    const float* W2 = (const float*)d_in[3];
    const float* W3 = (const float*)d_in[4];
    float* out = (float*)d_out;

    const size_t NW = (size_t)E_ * H_ * D_;        // 16,777,216 elems per W
    char* ws = (char*)d_ws;
    size_t off = 0;
    int*    cnt  = (int*)(ws + off);   off += 256;
    int*    rec  = (int*)(ws + off);   off += (size_t)E_ * TCAP * 4;
    float*  wsl  = (float*)(ws + off); off += (size_t)E_ * TCAP * 4;
    __bf16* xb   = (__bf16*)(ws + off); off += (size_t)T_ * D_ * 2;
    __bf16* W1b  = (__bf16*)(ws + off); off += NW * 2;
    __bf16* W2b  = (__bf16*)(ws + off); off += NW * 2;
    __bf16* W3b  = (__bf16*)(ws + off); off += NW * 2;
    __bf16* g    = (__bf16*)(ws + off); off += (size_t)(2 * T_ + BM) * H_ * 2;
    float*  y    = (float*)(ws + off);  off += (size_t)(2 * T_ + BM) * D_ * 4;
    // total off ~= 176.6 MB; requires ws_size >= off.

    hipLaunchKernelGGL(moe_init, dim3(64), dim3(256), 0, stream, cnt, rec, wsl);
    hipLaunchKernelGGL(cvt_f32_bf16, dim3((T_ * (size_t)D_) / 8 / 256), dim3(256),
                       0, stream, x, xb);
    hipLaunchKernelGGL(cvt_f32_bf16, dim3(NW / 8 / 256), dim3(256), 0, stream, W1, W1b);
    hipLaunchKernelGGL(cvt_f32_bf16, dim3(NW / 8 / 256), dim3(256), 0, stream, W2, W2b);
    hipLaunchKernelGGL(cvt_f32_bf16, dim3(NW / 8 / 256), dim3(256), 0, stream, W3, W3b);
    hipLaunchKernelGGL(moe_gate, dim3(T_ / 128), dim3(128), 0, stream,
                       x, Wg, cnt, rec, wsl);
    hipLaunchKernelGGL(moe_ffn1, dim3(E_ * (TCAP / BM), H_ / 128), dim3(BLK), 0,
                       stream, xb, W1b, W2b, cnt, rec, g);
    hipLaunchKernelGGL(moe_ffn2, dim3(E_ * (TCAP / BM), D_ / 256), dim3(BLK), 0,
                       stream, g, W3b, cnt, rec, wsl, y);
    hipLaunchKernelGGL(moe_combine, dim3(T_ * D_ / 4 / 256), dim3(256), 0,
                       stream, y, out);
}
